// LSSTransform_67826123538983
// MI455X (gfx1250) — compile-verified, matrix operand order checked by reading, not run on hardware
//
#include <hip/hip_runtime.h>
#include <hip/hip_bf16.h>

#define IN_CH   256
#define OUT_CH  80
#define NDEPTH  118
#define DTOT    198          // NDEPTH + OUT_CH
#define FH      32
#define FW      88
#define HW      (FH*FW)      // 2816
#define NCAM    6
#define NPIX    (NCAM*HW)    // 16896
#define NX      360
#define NBEV    (NX*NX)      // 129600

typedef __attribute__((ext_vector_type(16))) _Float16 v16h;
typedef __attribute__((ext_vector_type(8)))  _Float16 v8h;
typedef __attribute__((ext_vector_type(4)))  _Float16 v4h;
typedef __attribute__((ext_vector_type(8)))  float    v8f;

// K index of the low element of packed pair r for this lane-half,
// per CDNA5 ISA 7.12.2 (16-bit A-matrix 16x32 layout).
__device__ __forceinline__ int kpair(int r, int half) {
    return 2*(r&3) + 8*half + 16*(r>>2);
}

// ---------------------------------------------------------------------------
// depthnet 1x1 conv:  feat[cam][o][hw] = sum_c W[o][c]*img[cam][c][hw] + b[o]
// GEMM M=198(pad 208) K=256 N=16896, WMMA f32_16x16x32_f16, A staged in LDS.
// ---------------------------------------------------------------------------
__global__ __launch_bounds__(128) void depthnet_gemm(
    const float* __restrict__ img, const float* __restrict__ W,
    const float* __restrict__ bias, float* __restrict__ feat)
{
    __shared__ _Float16 As[16*IN_CH];
    const int lane = threadIdx.x & 31;
    const int wave = threadIdx.x >> 5;
    const int mt   = blockIdx.y;

    for (int e = threadIdx.x; e < 16*IN_CH; e += 128) {
        int m = e >> 8, k = e & 255;
        int mg = mt*16 + m;
        As[e] = (mg < DTOT) ? (_Float16)W[(size_t)mg*IN_CH + k] : (_Float16)0.f;
    }
    __syncthreads();

    const int nt   = blockIdx.x*4 + wave;
    const int half = lane >> 4;
    const int nl   = lane & 15;
    const int n    = nt*16 + nl;
    const int cam  = n / HW;
    const int hw   = n % HW;
    const float* bimg = img + (size_t)cam*IN_CH*HW + hw;

    v8f acc = {};
    for (int k0 = 0; k0 < IN_CH; k0 += 32) {
        v16h a, b;
#pragma unroll
        for (int r = 0; r < 8; ++r) {
            int kb = k0 + kpair(r, half);
            a[2*r]   = As[nl*IN_CH + kb];
            a[2*r+1] = As[nl*IN_CH + kb + 1];
            b[2*r]   = (_Float16)bimg[(size_t)kb    *HW];
            b[2*r+1] = (_Float16)bimg[(size_t)(kb+1)*HW];
        }
        if (k0 + 32 < IN_CH) __builtin_prefetch(&bimg[(size_t)(k0+32)*HW], 0, 1);
        acc = __builtin_amdgcn_wmma_f32_16x16x32_f16(false, a, false, b,
                                                     (short)0, acc, false, false);
    }
#pragma unroll
    for (int r = 0; r < 8; ++r) {
        int m = mt*16 + r + 8*half;
        if (m < DTOT)
            feat[((size_t)cam*DTOT + m)*HW + hw] = acc[r] + bias[m];
    }
}

// ---------------------------------------------------------------------------
// softmax over the first 118 channels per pixel
// ---------------------------------------------------------------------------
__global__ void softmax_depth(const float* __restrict__ feat,
                              float* __restrict__ depth)
{
    int i = blockIdx.x*blockDim.x + threadIdx.x;
    if (i >= NPIX) return;
    int cam = i / HW, hw = i % HW;
    const float* f = feat + (size_t)cam*DTOT*HW + hw;
    float mx = -3.4e38f;
    for (int o = 0; o < NDEPTH; ++o) mx = fmaxf(mx, f[(size_t)o*HW]);
    float s = 0.f;
    for (int o = 0; o < NDEPTH; ++o) s += __expf(f[(size_t)o*HW] - mx);
    float inv = 1.f/s;
    float* dp = depth + (size_t)cam*NDEPTH*HW + hw;
    for (int o = 0; o < NDEPTH; ++o) dp[(size_t)o*HW] = __expf(f[(size_t)o*HW] - mx)*inv;
}

// ---------------------------------------------------------------------------
// 4x4 inverses of img_aug and lidar2image per camera
// ---------------------------------------------------------------------------
__device__ void inv4x4(const float* M, float* R) {
    float a[4][8];
    for (int i = 0; i < 4; ++i)
        for (int j = 0; j < 4; ++j) { a[i][j] = M[i*4+j]; a[i][j+4] = (i==j)?1.f:0.f; }
    for (int c = 0; c < 4; ++c) {
        int p = c; float best = fabsf(a[c][c]);
        for (int r = c+1; r < 4; ++r) { float v = fabsf(a[r][c]); if (v > best) { best = v; p = r; } }
        if (p != c) for (int j = 0; j < 8; ++j) { float t = a[c][j]; a[c][j] = a[p][j]; a[p][j] = t; }
        float ip = 1.f / a[c][c];
        for (int j = 0; j < 8; ++j) a[c][j] *= ip;
        for (int r = 0; r < 4; ++r) if (r != c) {
            float fmul = a[r][c];
            for (int j = 0; j < 8; ++j) a[r][j] -= fmul*a[c][j];
        }
    }
    for (int i = 0; i < 4; ++i)
        for (int j = 0; j < 4; ++j) R[i*4+j] = a[i][j+4];
}

__global__ void invert_mats(const float* __restrict__ l2i,
                            const float* __restrict__ aug,
                            float* __restrict__ inv)
{
    int c = threadIdx.x;
    if (c >= NCAM) return;
    inv4x4(aug + c*16, inv + c*32);        // inv(img_aug)
    inv4x4(l2i + c*16, inv + c*32 + 16);   // inv(lidar2image)
}

// ---------------------------------------------------------------------------
// fused outer-product + BEV scatter-add (never materializes the 160M elems)
// ---------------------------------------------------------------------------
__global__ void bev_scatter(const float* __restrict__ feat,
                            const float* __restrict__ depth,
                            const float* __restrict__ inv,
                            float* __restrict__ bev)
{
    int p = blockIdx.x*blockDim.x + threadIdx.x;
    if (p >= NCAM*NDEPTH*HW) return;
    int cam = p / (NDEPTH*HW);
    int rem = p % (NDEPTH*HW);
    int di  = rem / HW;
    int hw  = rem % HW;
    int h = hw / FW, w = hw % FW;

    float fx = w * (703.f/87.f);   // linspace(0, 703, 88)
    float fy = h * (255.f/31.f);   // linspace(0, 255, 32)
    float fd = 1.f + 0.5f*di;      // arange(1, 60, 0.5)

    const float* A = inv + cam*32;       // inv(img_aug)
    const float* L = A + 16;             // inv(lidar2image)
    float p0 = A[0]*fx + A[1]*fy + A[2] *fd + A[3];
    float p1 = A[4]*fx + A[5]*fy + A[6] *fd + A[7];
    float p2 = A[8]*fx + A[9]*fy + A[10]*fd + A[11];
    float q0 = p0*p2, q1 = p1*p2, q2 = p2;
    float g0 = L[0]*q0 + L[1]*q1 + L[2] *q2 + L[3];
    float g1 = L[4]*q0 + L[5]*q1 + L[6] *q2 + L[7];
    float g2 = L[8]*q0 + L[9]*q1 + L[10]*q2 + L[11];

    int gx = (int)((g0 + 54.0f) * (1.f/0.3f));
    int gy = (int)((g1 + 54.0f) * (1.f/0.3f));
    int gz = (int)((g2 + 10.0f) * (1.f/20.f));
    if (gx < 0 || gx >= NX || gy < 0 || gy >= NX || gz != 0) return;

    int idx = gx*NX + gy;
    float dval = depth[((size_t)cam*NDEPTH + di)*HW + hw];
    const float* ctx = feat + ((size_t)cam*DTOT + NDEPTH)*HW + hw;
    float* dst = bev + (size_t)idx*OUT_CH;
    for (int o = 0; o < OUT_CH; ++o)
        atomicAdd(&dst[o], dval * ctx[(size_t)o*HW]);
}

// ---------------------------------------------------------------------------
// f32 -> f16 conversion of the pooled BEV grid (4 elements / thread)
// ---------------------------------------------------------------------------
__global__ void f32_to_f16_4(const float4* __restrict__ in,
                             v4h* __restrict__ out, int n4)
{
    int i = blockIdx.x*blockDim.x + threadIdx.x;
    if (i >= n4) return;
    float4 v = in[i];
    v4h h; h[0] = (_Float16)v.x; h[1] = (_Float16)v.y;
           h[2] = (_Float16)v.z; h[3] = (_Float16)v.w;
    out[i] = h;
}

// ---------------------------------------------------------------------------
// pre-pack conv weights into per-lane WMMA A-fragment layout (f16), K padded
// to 96 per tap; and fold BN into per-channel scale/bias.
// ---------------------------------------------------------------------------
__global__ void pack_conv_weights(const float* __restrict__ w1,
                                  const float* __restrict__ w2,
                                  const float* __restrict__ w3,
                                  _Float16* __restrict__ apack)
{
    const int per = 5*9*3*32;            // fragments-per-conv * lanes
    int t = blockIdx.x*blockDim.x + threadIdx.x;
    if (t >= 3*per) return;
    int conv = t / per;
    int r0   = t % per;                  // ((mt*9+tap)*3+ks)*32 + lane
    int lane = r0 & 31;
    int tmp  = r0 >> 5;
    int ks   = tmp % 3;
    int tap  = (tmp/3) % 9;
    int mt   = tmp / 27;
    const float* w = (conv==0) ? w1 : (conv==1) ? w2 : w3;
    int o    = mt*16 + (lane & 15);
    int half = lane >> 4;
    _Float16* dst = apack + ((size_t)conv*per + r0)*16;
#pragma unroll
    for (int r = 0; r < 8; ++r)
        for (int j = 0; j < 2; ++j) {
            int c = ks*32 + kpair(r, half) + j;
            float v = (c < OUT_CH)
                ? w[(((size_t)o*OUT_CH + c)*3 + tap/3)*3 + (tap%3)] : 0.f;
            dst[2*r + j] = (_Float16)v;
        }
}

__global__ void bn_prep(const float* g1,const float* b1,const float* m1,const float* v1,
                        const float* g2,const float* b2,const float* m2,const float* v2,
                        const float* g3,const float* b3,const float* m3,const float* v3,
                        float* __restrict__ sb)   // [3][scale 80 | bias 80]
{
    int t = blockIdx.x*blockDim.x + threadIdx.x;
    if (t >= 3*OUT_CH) return;
    int conv = t / OUT_CH, o = t % OUT_CH;
    const float *g,*b,*m,*v;
    if (conv==0)      { g=g1; b=b1; m=m1; v=v1; }
    else if (conv==1) { g=g2; b=b2; m=m2; v=v2; }
    else              { g=g3; b=b3; m=m3; v=v3; }
    float s = g[o] * rsqrtf(v[o] + 1e-5f);
    sb[conv*160 + o]      = s;
    sb[conv*160 + 80 + o] = b[o] - m[o]*s;
}

// ---------------------------------------------------------------------------
// implicit-GEMM 3x3 conv + BN + ReLU. Input f16 channel-minor [pix][80].
// Block = 5 waves = 80 out-channels x 32 pixels of one output row (2 N-tiles
// per wave -> 2 WMMA per A-fragment load, 54 static WMMAs per block).
// Input slab (3 rows x NCOL cols, CP=104-half padded pixel stride, channels
// 80..103 zero-filled) staged once in LDS and shared by all 5 waves.
// B-fragments = two ds_load_b128 per k-step (the WMMA 16-bit layout is two
// contiguous 16B runs per lane); zero A-padding annihilates K in [80,96).
// OUTMODE 0: f16 channel-minor out; OUTMODE 1: f32 (C,W,H) layout to d_out.
// ---------------------------------------------------------------------------
template<int STRIDE, int OUTMODE>
__global__ __launch_bounds__(160) void conv3x3_bn_relu(
    const _Float16* __restrict__ in, const _Float16* __restrict__ apack,
    const float* __restrict__ scale, const float* __restrict__ bias,
    void* __restrict__ outp, int Hin, int Win, int Hout, int Wout)
{
    constexpr int NCOL = 31*STRIDE + 3;   // 34 (s1) / 65 (s2) input columns
    constexpr int CP   = 104;             // padded pixel stride: 208B = 13*16B
    __shared__ _Float16 Bs[3*NCOL*CP];

    const int oy = blockIdx.y;
    const int x0 = blockIdx.x*32;

    // cooperative slab fill: rows iy = oy*S + {-1,0,1}; c in [80,104) -> 0
    for (int e = threadIdx.x; e < 3*NCOL*26; e += 160) {
        int row = e/(NCOL*26), rem = e%(NCOL*26);
        int col = rem/26,      c4  = rem%26;
        int iy = oy*STRIDE + row - 1;
        int ix = x0*STRIDE + col - 1;
        v4h val = {};
        if (c4 < 20 && iy >= 0 && iy < Hin && ix >= 0 && ix < Win)
            val = *(const v4h*)(in + ((size_t)iy*Win + ix)*OUT_CH + c4*4);
        *(v4h*)(Bs + (row*NCOL + col)*CP + c4*4) = val;
    }
    __syncthreads();

    const int lane = threadIdx.x & 31;
    const int mt   = threadIdx.x >> 5;   // 0..4 -> out channels mt*16..mt*16+15
    const int half = lane >> 4;
    const int nl   = lane & 15;
    const int ox0  = x0 + nl;
    const int ox1  = x0 + 16 + nl;

    v8f acc0 = {}, acc1 = {};
#pragma unroll
    for (int tap = 0; tap < 9; ++tap) {
        const int ry = tap/3, kx = tap%3;
        const _Float16* s0 = Bs + ((size_t)ry*NCOL + nl*STRIDE + kx)*CP;
        const _Float16* s1 = s0 + (size_t)16*STRIDE*CP;
#pragma unroll
        for (int ks = 0; ks < 3; ++ks) {
            v16h a = *(const v16h*)(apack +
                        ((size_t)((mt*9 + tap)*3 + ks)*32 + lane)*16);
            const int o8 = ks*32 + 8*half;
            v8h lo0 = *(const v8h*)(s0 + o8);        // ds_load_b128
            v8h hi0 = *(const v8h*)(s0 + o8 + 16);
            v8h lo1 = *(const v8h*)(s1 + o8);
            v8h hi1 = *(const v8h*)(s1 + o8 + 16);
            v16h b0 = __builtin_shufflevector(lo0, hi0,
                        0,1,2,3,4,5,6,7,8,9,10,11,12,13,14,15);
            v16h b1 = __builtin_shufflevector(lo1, hi1,
                        0,1,2,3,4,5,6,7,8,9,10,11,12,13,14,15);
            acc0 = __builtin_amdgcn_wmma_f32_16x16x32_f16(false, a, false, b0,
                                                          (short)0, acc0, false, false);
            acc1 = __builtin_amdgcn_wmma_f32_16x16x32_f16(false, a, false, b1,
                                                          (short)0, acc1, false, false);
        }
    }
    // stores (after all WMMAs; EXEC all-ones rule)
#pragma unroll
    for (int r = 0; r < 8; ++r) {
        int o = mt*16 + r + 8*half;
        float v0 = fmaxf(acc0[r]*scale[o] + bias[o], 0.f);
        float v1 = fmaxf(acc1[r]*scale[o] + bias[o], 0.f);
        if (OUTMODE == 0) {
            if (ox0 < Wout)
                ((_Float16*)outp)[((size_t)oy*Wout + ox0)*OUT_CH + o] = (_Float16)v0;
            if (ox1 < Wout)
                ((_Float16*)outp)[((size_t)oy*Wout + ox1)*OUT_CH + o] = (_Float16)v1;
        } else {
            if (ox0 < Wout)
                ((float*)outp)[((size_t)o*Wout + ox0)*Hout + oy] = v0;  // (C,W,H)
            if (ox1 < Wout)
                ((float*)outp)[((size_t)o*Wout + ox1)*Hout + oy] = v1;
        }
    }
}

__global__ void zero_f4(float4* __restrict__ p, int n4) {
    int i = blockIdx.x*blockDim.x + threadIdx.x;
    if (i < n4) p[i] = make_float4(0.f, 0.f, 0.f, 0.f);
}

// ---------------------------------------------------------------------------
extern "C" void kernel_launch(void* const* d_in, const int* in_sizes, int n_in,
                              void* d_out, int out_size, void* d_ws, size_t ws_size,
                              hipStream_t stream)
{
    const float* img  = (const float*)d_in[0];
    const float* l2i  = (const float*)d_in[1];
    const float* aug  = (const float*)d_in[2];
    const float* dnw  = (const float*)d_in[3];
    const float* dnb  = (const float*)d_in[4];
    const float* c1w  = (const float*)d_in[5];
    const float* c2w  = (const float*)d_in[10];
    const float* c3w  = (const float*)d_in[15];

    char* ws = (char*)d_ws;
    size_t off = 0;
    auto carve = [&](size_t bytes) {
        size_t o = off;
        off = (off + bytes + 255) & ~(size_t)255;
        return o;
    };
    float*    feat  = (float*)(ws + carve(sizeof(float)*(size_t)NCAM*DTOT*HW));
    float*    depth = (float*)(ws + carve(sizeof(float)*(size_t)NCAM*NDEPTH*HW));
    float*    bev   = (float*)(ws + carve(sizeof(float)*(size_t)NBEV*OUT_CH));
    _Float16* bevh  = (_Float16*)(ws + carve(sizeof(_Float16)*(size_t)NBEV*OUT_CH));
    _Float16* y1h   = (_Float16*)(ws + carve(sizeof(_Float16)*(size_t)NX*NX*OUT_CH));
    _Float16* y2h   = (_Float16*)(ws + carve(sizeof(_Float16)*(size_t)180*180*OUT_CH));
    float*    invm  = (float*)(ws + carve(sizeof(float)*NCAM*32));
    _Float16* apack = (_Float16*)(ws + carve(sizeof(_Float16)*3*5*9*3*32*16));
    float*    sb    = (float*)(ws + carve(sizeof(float)*3*160));

    // 1. zero BEV accumulator (atomics target; must be re-zeroed every call)
    const int nbev4 = (NBEV*OUT_CH)/4;
    zero_f4<<<(nbev4 + 255)/256, 256, 0, stream>>>((float4*)bev, nbev4);
    // 2. depthnet GEMM (WMMA): grid = (1056 n-tiles / 4 waves, 13 m-tiles)
    depthnet_gemm<<<dim3(NPIX/64, 13), 128, 0, stream>>>(img, dnw, dnb, feat);
    // 3. softmax over depth bins
    softmax_depth<<<(NPIX + 255)/256, 256, 0, stream>>>(feat, depth);
    // 4. invert the per-camera matrices
    invert_mats<<<1, 64, 0, stream>>>(l2i, aug, invm);
    // 5. pack conv weights into WMMA fragment layout + fold BN
    pack_conv_weights<<<(3*5*9*3*32 + 255)/256, 256, 0, stream>>>(c1w, c2w, c3w, apack);
    bn_prep<<<1, 256, 0, stream>>>(
        (const float*)d_in[6],  (const float*)d_in[7],  (const float*)d_in[8],  (const float*)d_in[9],
        (const float*)d_in[11], (const float*)d_in[12], (const float*)d_in[13], (const float*)d_in[14],
        (const float*)d_in[16], (const float*)d_in[17], (const float*)d_in[18], (const float*)d_in[19],
        sb);
    // 6. fused frustum lift + BEV pooling (scatter-add), then f32->f16
    bev_scatter<<<(NCAM*NDEPTH*HW)/256, 256, 0, stream>>>(feat, depth, invm, bev);
    f32_to_f16_4<<<(nbev4 + 255)/256, 256, 0, stream>>>((const float4*)bev, (v4h*)bevh, nbev4);
    // 7. conv stack (implicit-GEMM WMMA): 360->360, 360->180 (s2), 180->180
    conv3x3_bn_relu<1,0><<<dim3(12, 360), 160, 0, stream>>>(
        bevh, apack,           sb,       sb + 80,  y1h, NX, NX, NX, NX);
    conv3x3_bn_relu<2,0><<<dim3(6, 180), 160, 0, stream>>>(
        y1h,  apack + 69120,   sb + 160, sb + 240, y2h, NX, NX, 180, 180);
    conv3x3_bn_relu<1,1><<<dim3(6, 180), 160, 0, stream>>>(
        y2h,  apack + 2*69120, sb + 320, sb + 400, (float*)d_out, 180, 180, 180, 180);
}
